// Stage5_56212531970270
// MI455X (gfx1250) — matrix operand order repaired and checked by reading, not compiled, not run on hardware
//
#include <hip/hip_runtime.h>
#include <hip/hip_bf16.h>
#include <math.h>

// ---------------------------------------------------------------------------
// CDNA5 (gfx1250) WMMA types
// ---------------------------------------------------------------------------
typedef __attribute__((ext_vector_type(16))) __bf16 v16bf;
typedef __attribute__((ext_vector_type(8)))  __bf16 v8bf;
typedef __attribute__((ext_vector_type(8)))  float  v8f;

static __device__ __forceinline__ __bf16 f2bf(float f) {
  unsigned u = __builtin_bit_cast(unsigned, f);
  unsigned r = u + 0x7FFFu + ((u >> 16) & 1u);   // round-to-nearest-even
  unsigned short h = (unsigned short)(r >> 16);
  return __builtin_bit_cast(__bf16, h);
}

static __device__ __forceinline__ v16bf cat16(v8bf lo, v8bf hi) {
  return __builtin_shufflevector(lo, hi, 0,1,2,3,4,5,6,7,8,9,10,11,12,13,14,15);
}

static __device__ __forceinline__ v8f wmma_bf16(v16bf a, v16bf b, v8f c) {
  return __builtin_amdgcn_wmma_f32_16x16x32_bf16(false, a, false, b, (short)0, c,
                                                 false, false);
}

// Fragment loaders / wmma batches (arrays stay in registers after inlining)
static __device__ __forceinline__ void load_a2(v16bf dst[2],
                                               const __bf16* const Ap[2], int kb) {
  #pragma unroll
  for (int mi = 0; mi < 2; ++mi)
    dst[mi] = cat16(*(const v8bf*)(Ap[mi] + kb), *(const v8bf*)(Ap[mi] + kb + 16));
}
static __device__ __forceinline__ void load_b4(v16bf dst[4],
                                               const __bf16* const Bp[4], int kb) {
  #pragma unroll
  for (int j = 0; j < 4; ++j)
    dst[j] = cat16(*(const v8bf*)(Bp[j] + kb), *(const v8bf*)(Bp[j] + kb + 16));
}
static __device__ __forceinline__ void mm8(v8f acc[2][4], const v16bf a[2],
                                           const v16bf b[4]) {
  #pragma unroll
  for (int mi = 0; mi < 2; ++mi)
    #pragma unroll
    for (int j = 0; j < 4; ++j)
      acc[mi][j] = wmma_bf16(a[mi], b[j], acc[mi][j]);
}
static __device__ __forceinline__ void mm4(v8f acc[4], v16bf a, const v16bf b[4]) {
  #pragma unroll
  for (int j = 0; j < 4; ++j) acc[j] = wmma_bf16(a, b[j], acc[j]);
}

// ---------------------------------------------------------------------------
// Elementwise helpers
// ---------------------------------------------------------------------------
__global__ __launch_bounds__(256) void f32_to_bf16_kernel(
    const float* __restrict__ in, __bf16* __restrict__ out, int n) {
  int base = (blockIdx.x * 256 + threadIdx.x) * 4;
  if (base + 3 < n) {
    #pragma unroll
    for (int i = 0; i < 4; ++i) out[base + i] = f2bf(in[base + i]);
  }
}

__global__ __launch_bounds__(256) void zero_kernel(unsigned int* __restrict__ p, int nwords) {
  int i = blockIdx.x * 256 + threadIdx.x;
  if (i < nwords) p[i] = 0u;
}

// Embedding gather: row = t*B+b, out[row][e] = bf16(table[tokens[row]][e])
__global__ __launch_bounds__(256) void gather_emb_kernel(
    const int* __restrict__ tokens, const float* __restrict__ table,
    __bf16* __restrict__ out) {
  const int row = blockIdx.x;                 // 0..4095
  const int tok = tokens[row];
  const float* src = table + (size_t)tok * 1024;
  __bf16* dst = out + (size_t)row * 1024;
  for (int e = threadIdx.x; e < 1024; e += 256) dst[e] = f2bf(src[e]);
}

// ---------------------------------------------------------------------------
// WMMA GEMM: C[M,N] = A[M,K](bf16,row-major) * B[N,K]^T(bf16,row-major)
//            (+ bias1[n] + bias2[n] if non-null)
// Wave tile = 32x64 (2 A frags x 4 B frags -> 8 wmma / k-step).
// Block = 256 threads = 8 waves (2 M x 4 N) -> block tile 64 x 256.
// Grid: (N/256, M/64). K-loop is a 2-deep ping-pong unrolled by 64:
// no register rotation, ~1 full iteration of load->use distance.
// Requires K % 64 == 0 (holds: K = 1024 everywhere here).
// ---------------------------------------------------------------------------
__global__ __launch_bounds__(256) void gemm_bf16_wmma(
    const __bf16* __restrict__ A, const __bf16* __restrict__ Bm,
    float* __restrict__ C, const float* __restrict__ bias1,
    const float* __restrict__ bias2, int N, int K) {
  const unsigned lane = threadIdx.x & 31u;
  const unsigned half = lane >> 4;
  const unsigned w    = threadIdx.x >> 5;
  const int m0 = blockIdx.y * 64 + (int)(w >> 2) * 32;
  const int n0 = blockIdx.x * 256 + (int)(w & 3) * 64;

  v8f acc[2][4];
  #pragma unroll
  for (int mi = 0; mi < 2; ++mi)
    #pragma unroll
    for (int j = 0; j < 4; ++j) acc[mi][j] = (v8f){};

  const __bf16* Ap[2];
  Ap[0] = A + (size_t)(m0 + (int)(lane & 15)) * K + half * 8;
  Ap[1] = Ap[0] + (size_t)16 * K;
  const __bf16* Bp[4];
  #pragma unroll
  for (int j = 0; j < 4; ++j)
    Bp[j] = Bm + (size_t)(n0 + j * 16 + (int)(lane & 15)) * K + half * 8;

  // ping-pong fragment buffers
  v16bf a0[2], b0[4], a1[2], b1[4];
  load_a2(a0, Ap, 0);   load_b4(b0, Bp, 0);
  load_a2(a1, Ap, 32);  load_b4(b1, Bp, 32);

  for (int kb = 64; kb < K; kb += 64) {
    mm8(acc, a0, b0);
    load_a2(a0, Ap, kb);
    load_b4(b0, Bp, kb);
    __builtin_prefetch(Ap[0] + kb + 128, 0, 2);   // global_prefetch_b8 (near)
    __builtin_prefetch(Bp[0] + kb + 128, 0, 2);
    mm8(acc, a1, b1);
    load_a2(a1, Ap, kb + 32);
    load_b4(b1, Bp, kb + 32);
  }
  mm8(acc, a0, b0);
  mm8(acc, a1, b1);

  // ---- epilogue ----
  #pragma unroll
  for (int j = 0; j < 4; ++j) {
    const int n = n0 + j * 16 + (int)(lane & 15);
    float b = (bias1 ? bias1[n] : 0.f) + (bias2 ? bias2[n] : 0.f);
    #pragma unroll
    for (int mi = 0; mi < 2; ++mi) {
      #pragma unroll
      for (int r = 0; r < 8; ++r) {
        const int m = m0 + mi * 16 + (int)half * 8 + r;
        C[(size_t)m * N + n] = acc[mi][j][r] + b;
      }
    }
  }
}

// ---------------------------------------------------------------------------
// Fused LSTM step: g = h_in * W_hh^T (+ xp slice); gates; c,h update.
// Each wave computes a 16x16 tile of h: 4 gate accumulators (i,f,g,o),
// K-loop ping-pong-pipelined. Grid: 64 blocks x 256 threads.
// ---------------------------------------------------------------------------
__global__ __launch_bounds__(256) void lstm_step_kernel(
    const __bf16* __restrict__ h_in,   // [128 x 1024] bf16 (prev h)
    const __bf16* __restrict__ Whh,    // [4096 x 1024] bf16
    const float*  __restrict__ xp_t,   // [128 x 4096] f32 (precomputed x-projection)
    float*        __restrict__ c_state,// [128 x 1024] f32
    __bf16*       __restrict__ h_out,  // [128 x 1024] bf16 (next h, ping-pong)
    __bf16*       __restrict__ rnn_bf, // [128 x 1024] bf16 slice of rnn archive
    float*        __restrict__ rnn_f)  // [128 x 1024] f32 slice of d_out rnn
{
  const unsigned lane = threadIdx.x & 31u;
  const unsigned half = lane >> 4;
  const unsigned wid  = blockIdx.x * 8u + (threadIdx.x >> 5); // 0..511
  const int m0 = (int)(wid >> 6) * 16;   // batch tile
  const int u0 = (int)(wid & 63) * 16;   // hidden tile

  v8f acc[4];
  #pragma unroll
  for (int g = 0; g < 4; ++g) acc[g] = (v8f){};

  const __bf16* Aptr = h_in + (size_t)(m0 + (int)(lane & 15)) * 1024 + half * 8;
  const __bf16* Bp[4];
  #pragma unroll
  for (int g = 0; g < 4; ++g)
    Bp[g] = Whh + (size_t)(g * 1024 + u0 + (int)(lane & 15)) * 1024 + half * 8;

  v16bf a0, a1, b0[4], b1[4];
  a0 = cat16(*(const v8bf*)(Aptr), *(const v8bf*)(Aptr + 16));
  load_b4(b0, Bp, 0);
  a1 = cat16(*(const v8bf*)(Aptr + 32), *(const v8bf*)(Aptr + 48));
  load_b4(b1, Bp, 32);

  for (int kb = 64; kb < 1024; kb += 64) {
    mm4(acc, a0, b0);
    a0 = cat16(*(const v8bf*)(Aptr + kb), *(const v8bf*)(Aptr + kb + 16));
    load_b4(b0, Bp, kb);
    __builtin_prefetch(Bp[0] + kb + 128, 0, 2);
    mm4(acc, a1, b1);
    a1 = cat16(*(const v8bf*)(Aptr + kb + 32), *(const v8bf*)(Aptr + kb + 48));
    load_b4(b1, Bp, kb + 32);
  }
  mm4(acc, a0, b0);
  mm4(acc, a1, b1);

  const int u = u0 + (int)(lane & 15);
  #pragma unroll
  for (int r = 0; r < 8; ++r) {
    const int m = m0 + (int)half * 8 + r;
    const float* xr = xp_t + (size_t)m * 4096;
    float gi = acc[0][r] + xr[u];
    float gf = acc[1][r] + xr[1024 + u];
    float gg = acc[2][r] + xr[2048 + u];
    float go = acc[3][r] + xr[3072 + u];
    float is = 1.f / (1.f + expf(-gi));
    float fs = 1.f / (1.f + expf(-gf));
    float os = 1.f / (1.f + expf(-go));
    float gt = tanhf(gg);
    const size_t idx = (size_t)m * 1024 + u;
    float c = fs * c_state[idx] + is * gt;
    c_state[idx] = c;
    float h = os * tanhf(c);
    rnn_f[idx] = h;
    __bf16 hb = f2bf(h);
    h_out[idx] = hb;
    rnn_bf[idx] = hb;
  }
}

// ---------------------------------------------------------------------------
// Fused normalized-Bahdanau scores + mask + softmax + attn*context.
// One block per (qt, b). Wave w reduces score for kt = w, w+8, w+16, w+24.
// ---------------------------------------------------------------------------
__global__ __launch_bounds__(256) void attention_kernel(
    const float* __restrict__ qf,       // [(t*B+b) x 1024]
    const float* __restrict__ kf,       // [(kt*B+b) x 1024]
    const float* __restrict__ ctx,      // [32 x 128 x 1024] f32
    const float* __restrict__ v_att,    // [1024]
    const float* __restrict__ att_bias, // [1024]
    const float* __restrict__ norm_scalar,
    const int*   __restrict__ ctx_len,  // [128]
    float*       __restrict__ attn_out) // [(qt*B+b) x 1024]
{
  __shared__ float s_red[256];
  __shared__ float s_sc[32];
  const int tid = threadIdx.x;
  const int b   = blockIdx.x & 127;
  const int qt  = blockIdx.x >> 7;

  // ||v_att||
  float s = 0.f;
  for (int u = tid; u < 1024; u += 256) { float v = v_att[u]; s += v * v; }
  s_red[tid] = s;
  __syncthreads();
  for (int st = 128; st > 0; st >>= 1) {
    if (tid < st) s_red[tid] += s_red[tid + st];
    __syncthreads();
  }
  const float scale = norm_scalar[0] / sqrtf(s_red[0]);

  const float* qrow = qf + ((size_t)qt * 128 + b) * 1024;
  const unsigned w = tid >> 5, lane = tid & 31;
  for (int kt = (int)w; kt < 32; kt += 8) {
    const float* krow = kf + ((size_t)kt * 128 + b) * 1024;
    float acc = 0.f;
    for (int u = (int)lane; u < 1024; u += 32)
      acc += v_att[u] * tanhf(qrow[u] + krow[u] + att_bias[u]);
    for (int off = 16; off > 0; off >>= 1) acc += __shfl_xor(acc, off, 32);
    if (lane == 0) s_sc[kt] = acc * scale;
  }
  __syncthreads();

  // masked softmax over kt (first wave: 32 lanes <-> 32 keys)
  if (tid < 32) {
    const int L = ctx_len[b];
    float sc = (tid < L) ? s_sc[tid] : -65504.0f;
    float mx = sc;
    for (int off = 16; off > 0; off >>= 1) mx = fmaxf(mx, __shfl_xor(mx, off, 32));
    float e = expf(sc - mx);
    float sum = e;
    for (int off = 16; off > 0; off >>= 1) sum += __shfl_xor(sum, off, 32);
    s_sc[tid] = e / sum;
  }
  __syncthreads();

  float* orow = attn_out + ((size_t)qt * 128 + b) * 1024;
  for (int c = tid; c < 1024; c += 256) {
    float acc = 0.f;
    #pragma unroll 4
    for (int kt = 0; kt < 32; ++kt)
      acc += s_sc[kt] * ctx[((size_t)kt * 128 + b) * 1024 + c];
    orow[c] = acc;
  }
}

// ---------------------------------------------------------------------------
// Host-side orchestration
// ---------------------------------------------------------------------------
extern "C" void kernel_launch(void* const* d_in, const int* in_sizes, int n_in,
                              void* d_out, int out_size, void* d_ws, size_t ws_size,
                              hipStream_t stream) {
  (void)in_sizes; (void)n_in; (void)out_size; (void)ws_size;

  const int*   ctx_len     = (const int*)d_in[0];
  const int*   tokens      = (const int*)d_in[1];
  const float* context     = (const float*)d_in[2];
  const float* emb_table   = (const float*)d_in[3];
  const float* W_ih        = (const float*)d_in[4];
  const float* W_hh        = (const float*)d_in[5];
  const float* b_ih        = (const float*)d_in[6];
  const float* b_hh        = (const float*)d_in[7];
  const float* Wq          = (const float*)d_in[8];
  const float* Wk          = (const float*)d_in[9];
  const float* v_att       = (const float*)d_in[10];
  const float* att_bias    = (const float*)d_in[11];
  const float* norm_scalar = (const float*)d_in[12];

  constexpr int T = 32, B = 128, H = 1024, Cc = 1024, E = 1024, Tk = 32;
  constexpr size_t TB = (size_t)T * B;   // 4096 decode rows
  constexpr size_t KB_ = (size_t)Tk * B; // 4096 context rows

  // workspace carve-up (all 256B aligned; total ~147 MB)
  char* ws = (char*)d_ws;
  size_t off = 0;
  auto take = [&](size_t bytes) {
    char* p = ws + off;
    off += (bytes + 255) & ~(size_t)255;
    return p;
  };
  __bf16* Wih_bf = (__bf16*)take((size_t)4 * H * E * 2);
  __bf16* Whh_bf = (__bf16*)take((size_t)4 * H * H * 2);
  __bf16* Wq_bf  = (__bf16*)take((size_t)H * H * 2);
  __bf16* Wk_bf  = (__bf16*)take((size_t)H * Cc * 2);
  __bf16* emb_bf = (__bf16*)take(TB * E * 2);
  __bf16* ctx_bf = (__bf16*)take(KB_ * Cc * 2);
  __bf16* rnn_bf = (__bf16*)take(TB * H * 2);
  __bf16* hbuf   = (__bf16*)take((size_t)2 * B * H * 2);  // ping-pong h
  float*  c_state= (float*)take((size_t)B * H * 4);
  float*  xp     = (float*)take(TB * (size_t)4 * H * 4);
  float*  q_f    = (float*)take(TB * H * 4);
  float*  k_f    = (float*)take(KB_ * H * 4);

  float* out_attn = (float*)d_out;                       // (Tq,B,C)
  float* out_rnn  = out_attn + TB * Cc;                  // (T,B,H)

  // 1) precision staging: fp32 -> bf16 (weights, context), embedding gather
  f32_to_bf16_kernel<<<(4 * H * E) / 1024, 256, 0, stream>>>(W_ih, Wih_bf, 4 * H * E);
  f32_to_bf16_kernel<<<(4 * H * H) / 1024, 256, 0, stream>>>(W_hh, Whh_bf, 4 * H * H);
  f32_to_bf16_kernel<<<(H * H) / 1024, 256, 0, stream>>>(Wq, Wq_bf, H * H);
  f32_to_bf16_kernel<<<(H * Cc) / 1024, 256, 0, stream>>>(Wk, Wk_bf, H * Cc);
  f32_to_bf16_kernel<<<(int)(KB_ * Cc) / 1024, 256, 0, stream>>>(context, ctx_bf, (int)(KB_ * Cc));
  gather_emb_kernel<<<(int)TB, 256, 0, stream>>>(tokens, emb_table, emb_bf);

  // 2) xp = emb * W_ih^T + (b_ih + b_hh)   [4096 x 4096 x 1024]
  gemm_bf16_wmma<<<dim3((4 * H) / 256, (int)TB / 64), 256, 0, stream>>>(
      emb_bf, Wih_bf, xp, b_ih, b_hh, 4 * H, E);

  // 3) zero LSTM state (ws is poisoned by harness)
  zero_kernel<<<512, 256, 0, stream>>>((unsigned int*)hbuf, 2 * B * H / 2);
  zero_kernel<<<512, 256, 0, stream>>>((unsigned int*)c_state, B * H);

  // 4) 32 sequential fused LSTM steps (WMMA recurrent GEMM + gates)
  for (int t = 0; t < T; ++t) {
    const __bf16* hin = hbuf + (size_t)(t & 1) * B * H;
    __bf16* hout      = hbuf + (size_t)((t + 1) & 1) * B * H;
    lstm_step_kernel<<<64, 256, 0, stream>>>(
        hin, Whh_bf, xp + (size_t)t * B * 4 * H, c_state, hout,
        rnn_bf + (size_t)t * B * H, out_rnn + (size_t)t * B * H);
  }

  // 5) attention projections: q = rnn * Wq^T, k = ctx * Wk^T
  gemm_bf16_wmma<<<dim3(H / 256, (int)TB / 64), 256, 0, stream>>>(
      rnn_bf, Wq_bf, q_f, nullptr, nullptr, H, H);
  gemm_bf16_wmma<<<dim3(H / 256, (int)KB_ / 64), 256, 0, stream>>>(
      ctx_bf, Wk_bf, k_f, nullptr, nullptr, H, Cc);

  // 6) fused scores + mask + softmax + attention-weighted context
  attention_kernel<<<T * B, 256, 0, stream>>>(
      q_f, k_f, context, v_att, att_bias, norm_scalar, ctx_len, out_attn);
}
